// mamba_19052474925269
// MI455X (gfx1250) — matrix operand order repaired and checked by reading, not compiled, not run on hardware
//
#include <hip/hip_runtime.h>
#include <hip/hip_bf16.h>
#include <math.h>

#define L_SEQ   2048
#define D_MODEL 1024
#define D_INNER 2048
#define D_STATE 16
#define D_CONV  4
#define DT_RANK 64
#define N_OUT   256
#define DBC_N   (DT_RANK + 2 * D_STATE)   // 96

typedef __attribute__((ext_vector_type(16))) __bf16 v16bf;
typedef __attribute__((ext_vector_type(8)))  __bf16 v8bf;
typedef __attribute__((ext_vector_type(8)))  float  v8f;
typedef __attribute__((ext_vector_type(4)))  unsigned int u32x4;
typedef __attribute__((ext_vector_type(8)))  int i32x8;
typedef __attribute__((ext_vector_type(4)))  int i32x4;

#if defined(__gfx1250__) && __has_builtin(__builtin_amdgcn_tensor_load_to_lds)
#define HAVE_TDM 1
#endif

// ---------------------------------------------------------------- helpers
__device__ __forceinline__ float sigmoidf_(float x) { return 1.f / (1.f + __expf(-x)); }
__device__ __forceinline__ float siluf_(float x)    { return x * sigmoidf_(x); }
__device__ __forceinline__ float softplusf_(float x){ return (x > 20.f) ? x : log1pf(__expf(x)); }

// ---------------------------------------------------------------- fp32 -> bf16
__global__ void cvt_bf16_kernel(const float* __restrict__ src, __bf16* __restrict__ dst, int n) {
  int i = blockIdx.x * blockDim.x + threadIdx.x;
  if (i < n) dst[i] = (__bf16)src[i];
}

__global__ void cvt_cols_bf16_kernel(const float* __restrict__ src, __bf16* __restrict__ dst,
                                     int rows, int scols, int col0, int ncols) {
  int i = blockIdx.x * blockDim.x + threadIdx.x;
  if (i >= rows * ncols) return;
  int r = i / ncols, c = i - r * ncols;
  dst[i] = (__bf16)src[(size_t)r * scols + col0 + c];
}

// ---------------------------------------------------------------- TDM 2D tile load -> LDS
// Builds a Tensor DMA Descriptor (D#) per CDNA5 ISA 8.3-8.5 and issues
// TENSOR_LOAD_TO_LDS. OOB rows/cols (beyond remK/remRows) are zero-filled by HW.
__device__ __forceinline__ void tdm_load_tile_2d(const __bf16* gtile, unsigned ldsOff,
                                                 int remK, int remRows,
                                                 int tileK, int tileRows,
                                                 int rowStrideElems) {
#ifdef HAVE_TDM
  unsigned long long ga = (unsigned long long)(const void*)gtile;
  u32x4 g0;
  g0[0] = 1u;                                                   // count=1 (valid), user mode
  g0[1] = ldsOff;                                               // lds_addr (bytes)
  g0[2] = (unsigned)ga;                                         // global_addr[31:0]
  g0[3] = (unsigned)((ga >> 32) & 0x01ffffffu) | (2u << 30);    // global_addr[56:32] | type=2
  i32x8 g1;
  g1[0] = (int)(1u << 16);                                      // wg_mask=0, data_size=1 (2B)
  g1[1] = (int)(((unsigned)remK & 0xffffu) << 16);              // tensor_dim0[15:0]
  g1[2] = (int)((((unsigned)remK >> 16) & 0xffffu) |
                (((unsigned)remRows & 0xffffu) << 16));         // tensor_dim0[31:16] | tensor_dim1[15:0]
  g1[3] = (int)((((unsigned)remRows >> 16) & 0xffffu) |
                (((unsigned)tileK & 0xffffu) << 16));           // tensor_dim1[31:16] | tile_dim0
  g1[4] = (int)((unsigned)tileRows & 0xffffu);                  // tile_dim1, tile_dim2=0
  g1[5] = (int)(unsigned)rowStrideElems;                        // tensor_dim0_stride[31:0]
  g1[6] = 0;                                                    // stride[47:32] | dim1_stride lo
  g1[7] = 0;
  i32x4 z4 = {0, 0, 0, 0};
#if __clang_major__ >= 23
  i32x8 z8 = {0, 0, 0, 0, 0, 0, 0, 0};
  __builtin_amdgcn_tensor_load_to_lds(g0, g1, z4, z4, z8, 0);
#else
  __builtin_amdgcn_tensor_load_to_lds(g0, g1, z4, z4, 0);
#endif
#endif
}

#ifndef HAVE_TDM
// Fallback cooperative copy global -> LDS (rows x 32 bf16), zero-fill rows >= validRows.
__device__ __forceinline__ void coop_copy_tile(__bf16* dst, const __bf16* src, int ldsrc,
                                               int validRows, int k0, int tid) {
  for (int idx = tid; idx < 128 * 4; idx += 256) {
    int r = idx >> 2, c = (idx & 3) * 8;
    v8bf v;
    if (r < validRows) v = *(const v8bf*)(src + (size_t)r * ldsrc + k0 + c);
    else { for (int j = 0; j < 8; ++j) v[j] = (__bf16)0.f; }
    *(v8bf*)(dst + r * 32 + c) = v;
  }
}
#endif

// ---------------------------------------------------------------- WMMA fragment loads (from LDS tile, ld = 32)
// A fragment 16x32 bf16 (ISA 7.12.2): lane = m + 16*half;
//   elems 0..7 -> K = 8*half + j;  elems 8..15 -> K = 16 + 8*half + j
__device__ __forceinline__ v16bf load_a_frag(const __bf16* A, int mTile, int lane) {
  const int m = mTile + (lane & 15);
  const int half = lane >> 4;
  const __bf16* p = A + m * 32 + 8 * half;
  v8bf lo = *(const v8bf*)(p);
  v8bf hi = *(const v8bf*)(p + 16);
  v16bf r;
#pragma unroll
  for (int j = 0; j < 8; ++j) { r[j] = lo[j]; r[j + 8] = hi[j]; }
  return r;
}

// B fragment 32x16 bf16 from row-major W-tile (B = W^T): lane = n + 16*khalf;
//   elems 0..7 -> K = 8*khalf + j;  elems 8..15 -> K = 16 + 8*khalf + j
__device__ __forceinline__ v16bf load_b_frag(const __bf16* W, int nTile, int lane) {
  const int n = nTile + (lane & 15);
  const int khalf = lane >> 4;
  const __bf16* p = W + n * 32 + 8 * khalf;
  v8bf lo = *(const v8bf*)(p);
  v8bf hi = *(const v8bf*)(p + 16);
  v16bf r;
#pragma unroll
  for (int j = 0; j < 8; ++j) { r[j] = lo[j]; r[j + 8] = hi[j]; }
  return r;
}

// ---------------------------------------------------------------- GEMM: C[M,N] = A[M,K] * W[N,K]^T (+bias,+resid,act)
// 256 threads = 8 waves; wave tile 32x64 (2x4 WMMA accums); block tile 128x128.
// A/B K-slabs staged in LDS by the Tensor Data Mover (or coop copy fallback).
#define ACT_NONE 0
#define ACT_SOFTPLUS 1

__global__ __launch_bounds__(256)
void gemm_bf16_wmma_kernel(const __bf16* __restrict__ A, const __bf16* __restrict__ W,
                           float* __restrict__ C, int M, int N, int K,
                           const float* __restrict__ bias,
                           const float* __restrict__ resid, int act) {
  __shared__ __align__(16) __bf16 sA[128][32];
  __shared__ __align__(16) __bf16 sB[128][32];

  const int tid  = threadIdx.x;
  const int lane = tid & 31;
  const int wave = tid >> 5;                    // 0..7
  const int mBlock = blockIdx.x * 128;
  const int nBlock = blockIdx.y * 128;
  const int mW = (wave >> 1) * 32;              // 0,32,64,96
  const int nW = (wave & 1) * 64;               // 0,64
  const int nValid = (N - nBlock < 128) ? (N - nBlock) : 128;

#ifdef HAVE_TDM
  const unsigned ldsA = (unsigned)(unsigned long long)(const void*)&sA[0][0];
  const unsigned ldsB = (unsigned)(unsigned long long)(const void*)&sB[0][0];
#endif

  v8f acc[2][4] = {};
  for (int k0 = 0; k0 < K; k0 += 32) {
#ifdef HAVE_TDM
    if (wave == 0) {
      tdm_load_tile_2d(A + (size_t)mBlock * K + k0, ldsA, K - k0, M - mBlock, 32, 128, K);
      tdm_load_tile_2d(W + (size_t)nBlock * K + k0, ldsB, K - k0, nValid,     32, 128, K);
      __builtin_amdgcn_s_wait_tensorcnt(0);
    }
    __syncthreads();
#else
    coop_copy_tile(&sA[0][0], A + (size_t)mBlock * K, K, 128,    k0, tid);
    coop_copy_tile(&sB[0][0], W + (size_t)nBlock * K, K, nValid, k0, tid);
    __syncthreads();
#endif
    v16bf a0 = load_a_frag(&sA[0][0], mW,      lane);
    v16bf a1 = load_a_frag(&sA[0][0], mW + 16, lane);
    v16bf b0 = load_b_frag(&sB[0][0], nW,      lane);
    v16bf b1 = load_b_frag(&sB[0][0], nW + 16, lane);
    v16bf b2 = load_b_frag(&sB[0][0], nW + 32, lane);
    v16bf b3 = load_b_frag(&sB[0][0], nW + 48, lane);
    acc[0][0] = __builtin_amdgcn_wmma_f32_16x16x32_bf16(false, a0, false, b0, (short)0, acc[0][0], false, false);
    acc[0][1] = __builtin_amdgcn_wmma_f32_16x16x32_bf16(false, a0, false, b1, (short)0, acc[0][1], false, false);
    acc[0][2] = __builtin_amdgcn_wmma_f32_16x16x32_bf16(false, a0, false, b2, (short)0, acc[0][2], false, false);
    acc[0][3] = __builtin_amdgcn_wmma_f32_16x16x32_bf16(false, a0, false, b3, (short)0, acc[0][3], false, false);
    acc[1][0] = __builtin_amdgcn_wmma_f32_16x16x32_bf16(false, a1, false, b0, (short)0, acc[1][0], false, false);
    acc[1][1] = __builtin_amdgcn_wmma_f32_16x16x32_bf16(false, a1, false, b1, (short)0, acc[1][1], false, false);
    acc[1][2] = __builtin_amdgcn_wmma_f32_16x16x32_bf16(false, a1, false, b2, (short)0, acc[1][2], false, false);
    acc[1][3] = __builtin_amdgcn_wmma_f32_16x16x32_bf16(false, a1, false, b3, (short)0, acc[1][3], false, false);
    __syncthreads();
  }

  // C/D layout: VGPR r -> M = tile + 8*(lane>>4) + r, N = tile + (lane&15)
  const int half = lane >> 4;
#pragma unroll
  for (int i = 0; i < 2; ++i) {
#pragma unroll
    for (int j = 0; j < 4; ++j) {
      const int n = nBlock + nW + j * 16 + (lane & 15);
      if (n >= N) continue;
#pragma unroll
      for (int r = 0; r < 8; ++r) {
        const int m = mBlock + mW + i * 16 + 8 * half + r;
        float v = acc[i][j][r];
        if (bias)  v += bias[n];
        if (resid) v += resid[(size_t)m * N + n];
        if (act == ACT_SOFTPLUS) v = softplusf_(v);
        C[(size_t)m * N + n] = v;
      }
    }
  }
}

// ---------------------------------------------------------------- causal depthwise conv (k=4) + SiLU
__global__ void conv_silu_kernel(const float* __restrict__ xz,
                                 const float* __restrict__ convw,
                                 const float* __restrict__ convb,
                                 float* __restrict__ xc_f, __bf16* __restrict__ xc_b) {
  int i = blockIdx.x * blockDim.x + threadIdx.x;
  if (i >= L_SEQ * D_INNER) return;
  int l = i / D_INNER, c = i - l * D_INNER;
  float acc = convb[c];
#pragma unroll
  for (int k = 0; k < D_CONV; ++k) {
    int ls = l - (D_CONV - 1) + k;
    if (ls >= 0) acc += xz[(size_t)ls * (2 * D_INNER) + c] * convw[c * D_CONV + k];
  }
  float s = siluf_(acc);
  xc_f[i] = s;
  xc_b[i] = (__bf16)s;
}

// ---------------------------------------------------------------- selective scan
// 16 lanes = 16 states of one channel; 2 channels/wave; 16 channels/block.
__global__ __launch_bounds__(256)
void scan_kernel(const float* __restrict__ delta, const float* __restrict__ xc,
                 const float* __restrict__ dbc, const float* __restrict__ A_log,
                 float* __restrict__ ys) {
  const int s  = threadIdx.x & 15;
  const int ch = blockIdx.x * 16 + (threadIdx.x >> 4);
  const float Aneg = -__expf(A_log[(size_t)ch * D_STATE + s]);
  float h = 0.f;
  for (int l = 0; l < L_SEQ; ++l) {
    const float dl = delta[(size_t)l * D_INNER + ch];
    const float xv = xc[(size_t)l * D_INNER + ch];
    const float Bv = dbc[(size_t)l * DBC_N + DT_RANK + s];
    const float Cv = dbc[(size_t)l * DBC_N + DT_RANK + D_STATE + s];
    h = __expf(dl * Aneg) * h + dl * xv * Bv;
    float p = h * Cv;
    p += __shfl_xor(p, 1, 16);
    p += __shfl_xor(p, 2, 16);
    p += __shfl_xor(p, 4, 16);
    p += __shfl_xor(p, 8, 16);
    if (s == 0) ys[(size_t)l * D_INNER + ch] = p;
  }
}

// ---------------------------------------------------------------- y = (ys + D*xc) * silu(z) -> bf16
__global__ void ymix_kernel(const float* __restrict__ ys, const float* __restrict__ xc,
                            const float* __restrict__ Dp, const float* __restrict__ xz,
                            __bf16* __restrict__ y_b) {
  int i = blockIdx.x * blockDim.x + threadIdx.x;
  if (i >= L_SEQ * D_INNER) return;
  int l = i / D_INNER, c = i - l * D_INNER;
  float z = xz[(size_t)l * (2 * D_INNER) + D_INNER + c];
  float y = (ys[i] + Dp[c] * xc[i]) * siluf_(z);
  y_b[i] = (__bf16)y;
}

// ---------------------------------------------------------------- host
extern "C" void kernel_launch(void* const* d_in, const int* in_sizes, int n_in,
                              void* d_out, int out_size, void* d_ws, size_t ws_size,
                              hipStream_t stream) {
  (void)in_sizes; (void)n_in; (void)out_size; (void)ws_size;
  const float* x     = (const float*)d_in[0];
  const float* W_in  = (const float*)d_in[1];
  const float* convw = (const float*)d_in[2];
  const float* convb = (const float*)d_in[3];
  const float* W_xp  = (const float*)d_in[4];
  const float* W_dt  = (const float*)d_in[5];
  const float* b_dt  = (const float*)d_in[6];
  const float* A_log = (const float*)d_in[7];
  const float* Dp    = (const float*)d_in[8];
  const float* W_out = (const float*)d_in[9];
  const float* W_lin = (const float*)d_in[10];
  const float* b_lin = (const float*)d_in[11];
  float* out = (float*)d_out;

  char* wp = (char*)d_ws;
  auto alloc = [&](size_t bytes) -> void* {
    void* r = (void*)wp;
    wp += (bytes + 255) & ~(size_t)255;
    return r;
  };
  __bf16* x_b    = (__bf16*)alloc((size_t)L_SEQ * D_MODEL * 2);
  __bf16* Win_b  = (__bf16*)alloc((size_t)2 * D_INNER * D_MODEL * 2);
  __bf16* Wxp_b  = (__bf16*)alloc((size_t)DBC_N * D_INNER * 2);
  __bf16* Wdt_b  = (__bf16*)alloc((size_t)D_INNER * DT_RANK * 2);
  __bf16* Wout_b = (__bf16*)alloc((size_t)D_MODEL * D_INNER * 2);
  __bf16* Wlin_b = (__bf16*)alloc((size_t)N_OUT * D_MODEL * 2);
  float*  xz     = (float*) alloc((size_t)L_SEQ * 2 * D_INNER * 4);
  float*  xc_f   = (float*) alloc((size_t)L_SEQ * D_INNER * 4);
  __bf16* xc_b   = (__bf16*)alloc((size_t)L_SEQ * D_INNER * 2);
  float*  dbc    = (float*) alloc((size_t)L_SEQ * DBC_N * 4);
  __bf16* dt_b   = (__bf16*)alloc((size_t)L_SEQ * DT_RANK * 2);
  float*  delta  = (float*) alloc((size_t)L_SEQ * D_INNER * 4);
  float*  ysb    = (float*) alloc((size_t)L_SEQ * D_INNER * 4);
  __bf16* y_b    = (__bf16*)alloc((size_t)L_SEQ * D_INNER * 2);
  float*  h_f    = (float*) alloc((size_t)L_SEQ * D_MODEL * 4);
  __bf16* h_b    = (__bf16*)alloc((size_t)L_SEQ * D_MODEL * 2);

  auto cvt = [&](const float* s, __bf16* d, int n) {
    cvt_bf16_kernel<<<(n + 255) / 256, 256, 0, stream>>>(s, d, n);
  };
  auto gemm = [&](const __bf16* A, const __bf16* W, float* C, int M, int N, int K,
                  const float* bias, const float* resid, int act) {
    dim3 grid(M / 128, (N + 127) / 128, 1);
    gemm_bf16_wmma_kernel<<<grid, 256, 0, stream>>>(A, W, C, M, N, K, bias, resid, act);
  };

  cvt(x,     x_b,    L_SEQ * D_MODEL);
  cvt(W_in,  Win_b,  2 * D_INNER * D_MODEL);
  cvt(W_xp,  Wxp_b,  DBC_N * D_INNER);
  cvt(W_dt,  Wdt_b,  D_INNER * DT_RANK);
  cvt(W_out, Wout_b, D_MODEL * D_INNER);
  cvt(W_lin, Wlin_b, N_OUT * D_MODEL);

  // 1) xz = x @ W_in^T                         [2048, 4096]
  gemm(x_b, Win_b, xz, L_SEQ, 2 * D_INNER, D_MODEL, nullptr, nullptr, ACT_NONE);

  // 2) depthwise causal conv + SiLU -> xc
  conv_silu_kernel<<<(L_SEQ * D_INNER + 255) / 256, 256, 0, stream>>>(xz, convw, convb, xc_f, xc_b);

  // 3) dbc = xc @ W_xproj^T                    [2048, 96]
  gemm(xc_b, Wxp_b, dbc, L_SEQ, DBC_N, D_INNER, nullptr, nullptr, ACT_NONE);

  // 4) dt slice -> bf16                        [2048, 64]
  cvt_cols_bf16_kernel<<<(L_SEQ * DT_RANK + 255) / 256, 256, 0, stream>>>(
      dbc, dt_b, L_SEQ, DBC_N, 0, DT_RANK);

  // 5) delta = softplus(dt @ W_dt^T + b_dt)    [2048, 2048]
  gemm(dt_b, Wdt_b, delta, L_SEQ, D_INNER, DT_RANK, b_dt, nullptr, ACT_SOFTPLUS);

  // 6) selective scan -> ys                    [2048, 2048]
  scan_kernel<<<D_INNER / 16, 256, 0, stream>>>(delta, xc_f, dbc, A_log, ysb);

  // 7) y = (ys + D*xc) * silu(z) -> bf16
  ymix_kernel<<<(L_SEQ * D_INNER + 255) / 256, 256, 0, stream>>>(ysb, xc_f, Dp, xz, y_b);

  // 8) h = y @ W_out^T + x (residual)          [2048, 1024]
  gemm(y_b, Wout_b, h_f, L_SEQ, D_MODEL, D_INNER, nullptr, x, ACT_NONE);
  cvt(h_f, h_b, L_SEQ * D_MODEL);

  // 9) out = h @ W_lin^T + b_lin               [2048, 256]
  gemm(h_b, Wlin_b, out, L_SEQ, N_OUT, D_MODEL, b_lin, nullptr, ACT_NONE);
}
